// GAN_24850680774936
// MI455X (gfx1250) — matrix operand order repaired and checked by reading, not compiled, not run on hardware
//
#include <hip/hip_runtime.h>
#include <hip/hip_bf16.h>
#include <math.h>

// Problem constants (from reference)
#define MEM_SIZE 262144
#define KEY_DIM  256
#define BATCH    128
#define CHOOSE_K 128
#define EPS      1e-8f

// ---------- types ----------
typedef __attribute__((ext_vector_type(16))) __bf16 v16bf;
typedef __attribute__((ext_vector_type(8)))  float  v8f;

// ---------- workspace layout (bytes). Requires ws_size >= ~130 MB ----------
#define WS_QN    0              // 128*256 f32  (normalized q)
#define WS_QBF   131072         // 128*256 bf16 (normalized q, bf16)
#define WS_TOP1  196608         // 128 int
#define WS_CORR  197120         // 128 int
#define WS_OLD   197632         // 128 int
#define WS_SIM   1048576        // 128*262144 f32 = 134,217,728 B

// ---------- helpers ----------
__device__ __forceinline__ unsigned short f2bf(float x) {
  unsigned u = __float_as_uint(x);
  unsigned r = (u + 0x7FFFu + ((u >> 16) & 1u)) >> 16;   // RNE
  return (unsigned short)r;
}
__device__ __forceinline__ unsigned pack2(float a, float b) {
  return (unsigned)f2bf(a) | ((unsigned)f2bf(b) << 16);
}
// monotone float<->u32 key (larger key == larger float)
__device__ __forceinline__ unsigned okey(float f) {
  unsigned u = __float_as_uint(f);
  return (u & 0x80000000u) ? ~u : (u | 0x80000000u);
}
__device__ __forceinline__ float fromkey(unsigned k) {
  unsigned u = (k & 0x80000000u) ? (k & 0x7fffffffu) : ~k;
  return __uint_as_float(u);
}

// ================= K1: normalize q rows, emit f32 + bf16 =================
__global__ __launch_bounds__(256) void k1_norm_q(const float* __restrict__ q_raw,
                                                 float* __restrict__ qn,
                                                 unsigned short* __restrict__ qbf) {
  const int wave = threadIdx.x >> 5, lane = threadIdx.x & 31;
  for (int row = wave; row < BATCH; row += 8) {
    const float* src = q_raw + row * KEY_DIM;
    float v[8]; float s = 0.f;
    #pragma unroll
    for (int j = 0; j < 8; ++j) { v[j] = src[lane + 32 * j]; s += v[j] * v[j]; }
    #pragma unroll
    for (int off = 16; off > 0; off >>= 1) s += __shfl_xor(s, off, 32);
    const float sc = 1.0f / fmaxf(sqrtf(s), EPS);
    #pragma unroll
    for (int j = 0; j < 8; ++j) {
      const float o = v[j] * sc;
      qn [row * KEY_DIM + lane + 32 * j] = o;
      qbf[row * KEY_DIM + lane + 32 * j] = f2bf(o);
    }
  }
}

// ================= K2: sim = qn @ mem_keys^T via bf16 WMMA =================
// Block: 256 threads = 8 waves, handles a 256-row slab of mem_keys.
// LDS: 256 rows x 256 bf16, row stride padded to 528 B (132 dwords) to break
// bank conflicts; total 135,168 B dynamic LDS (WGP budget is 320 KB).
#define LDS_ROW_U32 132
__global__ __launch_bounds__(256) void k2_sim(const float* __restrict__ mem_keys,
                                              const unsigned short* __restrict__ qbf,
                                              float* __restrict__ sim) {
  extern __shared__ unsigned int lds[];
  const int tid   = threadIdx.x;
  const int nbase = blockIdx.x * 256;

  // Stage slab: one mem_keys row per thread, f32 -> packed bf16 into LDS.
  {
    const float4* src = (const float4*)(mem_keys + (size_t)(nbase + tid) * KEY_DIM);
    unsigned int* dst = lds + tid * LDS_ROW_U32;
    #pragma unroll 4
    for (int j = 0; j < 64; ++j) {
      const float4 f = src[j];
      dst[2 * j]     = pack2(f.x, f.y);
      dst[2 * j + 1] = pack2(f.z, f.w);
    }
  }
  __syncthreads();

  const int wave = tid >> 5, lane = tid & 31;
  const int mrow = lane & 15, half = lane >> 4;

  // Preload A fragments (this wave's 16 query rows, all 8 k-steps).
  // CDNA5 16-bit A 16x32 layout: lane holds two contiguous 8-elt K-chunks at
  // K = 8*half and K = 8*half + 16  ->  two b128 loads per fragment.
  v16bf afrag[8];
  {
    const unsigned short* qrow = qbf + (size_t)(wave * 16 + mrow) * KEY_DIM;
    #pragma unroll
    for (int ks = 0; ks < 8; ++ks) {
      const uint4 c0 = *(const uint4*)(qrow + ks * 32 + 8 * half);
      const uint4 c1 = *(const uint4*)(qrow + ks * 32 + 8 * half + 16);
      v16bf a; ((uint4*)&a)[0] = c0; ((uint4*)&a)[1] = c1;
      afrag[ks] = a;
    }
  }

  // Each wave: its 16-row M-tile against all 16 N-tiles of the slab.
  for (int nt = 0; nt < 16; ++nt) {
    v8f acc = {};
    const int brow = nt * 16 + mrow;      // B lane: n = lane&15
    #pragma unroll
    for (int ks = 0; ks < 8; ++ks) {
      // B 32x16 layout: lane holds K = 8*half .. +7 and K = 8*half + 16 .. +7 of column n.
      const unsigned int* bp = lds + brow * LDS_ROW_U32 + (ks * 16 + 4 * half);
      const uint4 c0 = *(const uint4*)(bp);
      const uint4 c1 = *(const uint4*)(bp + 8);
      v16bf b; ((uint4*)&b)[0] = c0; ((uint4*)&b)[1] = c1;
      acc = __builtin_amdgcn_wmma_f32_16x16x32_bf16(
          false, afrag[ks], false, b, (short)0, acc, false, false);
    }
    // C layout: VGPR r -> M = r + 8*half, N = lane&15.
    const int col = nbase + nt * 16 + mrow;
    const int mg0 = wave * 16 + 8 * half;
    #pragma unroll
    for (int r = 0; r < 8; ++r)
      sim[(size_t)(mg0 + r) * MEM_SIZE + col] = acc[r];
  }
}

// ============ K3: per-query exact top-128 select + softmax posterior ============
// One block per query. Two-level 16-bit radix select via 65536-bin LDS histogram
// (256 KB dynamic LDS). Ties at the threshold resolved lowest-index-first
// (matches jax.lax.top_k stability) via per-thread-chunk prefix counts.
__global__ __launch_bounds__(256) void k3_topk(const float* __restrict__ sim,
                                               const int* __restrict__ mem_values,
                                               const int* __restrict__ label,
                                               float* __restrict__ post_out,
                                               int* __restrict__ top1_out,
                                               int* __restrict__ corr_out) {
  extern __shared__ unsigned int sh[];            // 65536 bins
  __shared__ unsigned long long best[256];
  __shared__ unsigned int scnt[256], spref[256];
  __shared__ int sh_hib, sh_ghi, sh_need, sh_E;
  __shared__ unsigned sh_T;
  __shared__ float sh_S, sh_N, sh_max;
  __shared__ int sh_tielab;

  const int q = blockIdx.x, tid = threadIdx.x;
  const float* row = sim + (size_t)q * MEM_SIZE;

  // ---- pass A: high-16 histogram + argmax ----
  for (int i = tid; i < 65536; i += 256) sh[i] = 0;
  __syncthreads();
  unsigned long long mybest = 0ull;
  for (int i = tid; i < MEM_SIZE; i += 256) {
    const unsigned key = okey(row[i]);
    atomicAdd(&sh[key >> 16], 1u);
    const unsigned long long cand =
        ((unsigned long long)key << 32) | (unsigned)(~i);
    if (cand > mybest) mybest = cand;
  }
  best[tid] = mybest;
  __syncthreads();
  for (int s = 128; s > 0; s >>= 1) {
    if (tid < s && best[tid + s] > best[tid]) best[tid] = best[tid + s];
    __syncthreads();
  }
  if (tid == 0) {
    int cum = 0;
    for (int b = 65535; b >= 0; --b) {
      const int c = (int)sh[b];
      if (cum + c >= CHOOSE_K) { sh_hib = b; sh_ghi = cum; break; }
      cum += c;
    }
    sh_need = CHOOSE_K - sh_ghi;
    sh_max  = fromkey((unsigned)(best[0] >> 32));
    top1_out[q] = (int)~((unsigned)(best[0] & 0xffffffffull));
  }
  __syncthreads();

  // ---- pass B: low-16 histogram within threshold bin ----
  const int hib = sh_hib;
  for (int i = tid; i < 65536; i += 256) sh[i] = 0;
  __syncthreads();
  for (int i = tid; i < MEM_SIZE; i += 256) {
    const unsigned key = okey(row[i]);
    if ((int)(key >> 16) == hib) atomicAdd(&sh[key & 0xffffu], 1u);
  }
  __syncthreads();
  if (tid == 0) {
    int cum = 0;
    const int need = sh_need;
    for (int b = 65535; b >= 0; --b) {
      const int c = (int)sh[b];
      if (cum + c >= need) {
        sh_T = ((unsigned)hib << 16) | (unsigned)b;
        sh_E = need - cum;                 // equals to take (lowest indices)
        break;
      }
      cum += c;
    }
    sh_S = 0.f; sh_N = 0.f; sh_tielab = 0;
  }
  __syncthreads();

  // ---- pass C: softmax accumulation over key > T, plus E lowest-index ties ----
  const unsigned T = sh_T;
  const int E = sh_E;
  const float mx = sh_max;
  const int c0 = tid * (MEM_SIZE / 256), c1 = c0 + (MEM_SIZE / 256);
  float S = 0.f, N = 0.f; int eq = 0;
  for (int i = c0; i < c1; ++i) {
    const unsigned key = okey(row[i]);
    if (key > T) {
      const float e = expf(row[i] - mx);
      S += e;
      if (mem_values[i] == 1) N += e;
    } else if (key == T) ++eq;
  }
  scnt[tid] = (unsigned)eq;
  __syncthreads();
  if (tid == 0) { unsigned p = 0; for (int t = 0; t < 256; ++t) { spref[t] = p; p += scnt[t]; } }
  __syncthreads();
  {
    const unsigned base = spref[tid];
    int r = 0;
    if (base < (unsigned)E) {
      for (int i = c0; i < c1; ++i) {
        if (okey(row[i]) == T) {
          if (base + (unsigned)r < (unsigned)E && mem_values[i] == 1)
            atomicAdd(&sh_tielab, 1);
          ++r;
        }
      }
    }
  }
  atomicAdd(&sh_S, S);
  atomicAdd(&sh_N, N);
  __syncthreads();
  if (tid == 0) {
    const float eT = expf(fromkey(T) - mx);
    const float Ss = sh_S + (float)E * eT;
    const float Ns = sh_N + (float)sh_tielab * eT;
    float p = Ns / Ss;
    p = fminf(fmaxf(p, EPS), 1.0f - EPS);
    post_out[q] = p;
    const int t1 = top1_out[q];
    corr_out[q] = (mem_values[t1] == label[q]) ? 1 : 0;
  }
}

// ============ K4: oldest = indices of 128 largest ages, jax top_k order ============
__global__ __launch_bounds__(256) void k4_oldest(const int* __restrict__ mem_age,
                                                 int* __restrict__ oldest) {
  extern __shared__ unsigned int sh[];            // 65536 bins
  __shared__ unsigned long long list[BATCH];
  __shared__ unsigned int scnt[256], spref[256], gpos;
  __shared__ int sh_hib, sh_ghi, sh_need, sh_G, sh_E;
  __shared__ unsigned sh_T;
  const int tid = threadIdx.x;

  for (int i = tid; i < 65536; i += 256) sh[i] = 0;
  if (tid == 0) gpos = 0;
  __syncthreads();
  for (int i = tid; i < MEM_SIZE; i += 256) {
    const unsigned k = (unsigned)mem_age[i] ^ 0x80000000u;
    atomicAdd(&sh[k >> 16], 1u);
  }
  __syncthreads();
  if (tid == 0) {
    int cum = 0;
    for (int b = 65535; b >= 0; --b) {
      const int c = (int)sh[b];
      if (cum + c >= BATCH) { sh_hib = b; sh_ghi = cum; break; }
      cum += c;
    }
    sh_need = BATCH - sh_ghi;
  }
  __syncthreads();
  const int hib = sh_hib;
  for (int i = tid; i < 65536; i += 256) sh[i] = 0;
  __syncthreads();
  for (int i = tid; i < MEM_SIZE; i += 256) {
    const unsigned k = (unsigned)mem_age[i] ^ 0x80000000u;
    if ((int)(k >> 16) == hib) atomicAdd(&sh[k & 0xffffu], 1u);
  }
  __syncthreads();
  if (tid == 0) {
    int cum = 0;
    const int need = sh_need;
    for (int b = 65535; b >= 0; --b) {
      const int c = (int)sh[b];
      if (cum + c >= need) {
        sh_T = ((unsigned)hib << 16) | (unsigned)b;
        sh_G = sh_ghi + cum;
        sh_E = need - cum;
        break;
      }
      cum += c;
    }
  }
  __syncthreads();
  const unsigned T = sh_T;
  const int E = sh_E, G = sh_G;
  const int c0 = tid * (MEM_SIZE / 256), c1 = c0 + (MEM_SIZE / 256);
  int eq = 0;
  for (int i = c0; i < c1; ++i) {
    const unsigned k = (unsigned)mem_age[i] ^ 0x80000000u;
    if (k > T) {
      const unsigned p = atomicAdd(&gpos, 1u);
      list[p] = ((unsigned long long)k << 32) | (unsigned)(~i);
    } else if (k == T) ++eq;
  }
  scnt[tid] = (unsigned)eq;
  __syncthreads();
  if (tid == 0) { unsigned p = 0; for (int t = 0; t < 256; ++t) { spref[t] = p; p += scnt[t]; } }
  __syncthreads();
  {
    const unsigned base = spref[tid];
    int r = 0;
    if (base < (unsigned)E) {
      for (int i = c0; i < c1; ++i) {
        const unsigned k = (unsigned)mem_age[i] ^ 0x80000000u;
        if (k == T) {
          if (base + (unsigned)r < (unsigned)E)
            list[G + base + r] = ((unsigned long long)k << 32) | (unsigned)(~i);
          ++r;
        }
      }
    }
  }
  __syncthreads();
  // bitonic sort the 128 (agekey, ~idx) entries, descending
  for (int k2 = 2; k2 <= BATCH; k2 <<= 1) {
    for (int j = k2 >> 1; j > 0; j >>= 1) {
      const int i = tid;
      if (i < BATCH) {
        const int ixj = i ^ j;
        if (ixj > i) {
          const bool desc = ((i & k2) == 0);
          const unsigned long long a = list[i], b = list[ixj];
          if (desc ? (a < b) : (a > b)) { list[i] = b; list[ixj] = a; }
        }
      }
      __syncthreads();
    }
  }
  if (tid < BATCH) oldest[tid] = (int)~((unsigned)(list[tid] & 0xffffffffull));
}

// ============ K5: bulk output initialization (pure streaming) ============
__global__ void k5_copy_keys(const float4* __restrict__ src, float4* __restrict__ dst) {
  const size_t i = (size_t)blockIdx.x * blockDim.x + threadIdx.x;
  dst[i] = src[i];
}
__global__ void k5_copy_meta(const int* __restrict__ vals, const int* __restrict__ age,
                             float* __restrict__ ovals, float* __restrict__ oage) {
  const int i = blockIdx.x * blockDim.x + threadIdx.x;
  ovals[i] = (float)vals[i];
  oage[i]  = (float)(age[i] + 1);
}

// ============ K6: serialized scatter of the 128 memory writes ============
__global__ __launch_bounds__(256) void k6_apply(const float* __restrict__ qn,
                                                const float* __restrict__ mem_keys,
                                                const int* __restrict__ label,
                                                const int* __restrict__ top1,
                                                const int* __restrict__ corr,
                                                const int* __restrict__ oldest,
                                                float* __restrict__ okeys,
                                                float* __restrict__ ovals,
                                                float* __restrict__ oage) {
  __shared__ float red[256];
  __shared__ float s_scale;
  const int t = threadIdx.x;
  for (int i = 0; i < BATCH; ++i) {
    const int c = corr[i];                      // uniform across block
    const int idx = c ? top1[i] : oldest[i];
    float w;
    if (c) {
      const float v = qn[i * KEY_DIM + t] + mem_keys[(size_t)top1[i] * KEY_DIM + t];
      red[t] = v * v;
      __syncthreads();
      for (int s = 128; s > 0; s >>= 1) {
        if (t < s) red[t] += red[t + s];
        __syncthreads();
      }
      if (t == 0) s_scale = 1.0f / fmaxf(sqrtf(red[0]), EPS);
      __syncthreads();
      w = v * s_scale;
    } else {
      w = qn[i * KEY_DIM + t];
    }
    okeys[(size_t)idx * KEY_DIM + t] = w;
    if (t == 0) { ovals[idx] = (float)label[i]; oage[idx] = 0.0f; }
    __syncthreads();                            // enforce i-order (last wins)
  }
}

// ============================ launch ============================
extern "C" void kernel_launch(void* const* d_in, const int* in_sizes, int n_in,
                              void* d_out, int out_size, void* d_ws, size_t ws_size,
                              hipStream_t stream) {
  (void)in_sizes; (void)n_in; (void)out_size; (void)ws_size;
  const float* q_raw      = (const float*)d_in[0];
  const int*   label      = (const int*)  d_in[1];
  const float* mem_keys   = (const float*)d_in[2];
  const int*   mem_values = (const int*)  d_in[3];
  const int*   mem_age    = (const int*)  d_in[4];

  char* ws = (char*)d_ws;
  float*          qn     = (float*)(ws + WS_QN);
  unsigned short* qbf    = (unsigned short*)(ws + WS_QBF);
  int*            top1   = (int*)(ws + WS_TOP1);
  int*            corr   = (int*)(ws + WS_CORR);
  int*            oldest = (int*)(ws + WS_OLD);
  float*          sim    = (float*)(ws + WS_SIM);   // needs ~129 MB workspace

  float* out    = (float*)d_out;
  float* o_post = out;
  float* o_keys = out + BATCH;
  float* o_vals = o_keys + (size_t)MEM_SIZE * KEY_DIM;
  float* o_age  = o_vals + MEM_SIZE;

  hipLaunchKernelGGL(k1_norm_q, dim3(1), dim3(256), 0, stream, q_raw, qn, qbf);
  hipLaunchKernelGGL(k2_sim, dim3(MEM_SIZE / 256), dim3(256),
                     256 * 528 /* 135168 B LDS */, stream, mem_keys, qbf, sim);
  hipLaunchKernelGGL(k3_topk, dim3(BATCH), dim3(256),
                     65536 * 4 /* 256 KB LDS */, stream,
                     sim, mem_values, label, o_post, top1, corr);
  hipLaunchKernelGGL(k4_oldest, dim3(1), dim3(256),
                     65536 * 4, stream, mem_age, oldest);
  hipLaunchKernelGGL(k5_copy_keys, dim3((MEM_SIZE * KEY_DIM / 4) / 256), dim3(256),
                     0, stream, (const float4*)mem_keys, (float4*)o_keys);
  hipLaunchKernelGGL(k5_copy_meta, dim3(MEM_SIZE / 256), dim3(256),
                     0, stream, mem_values, mem_age, o_vals, o_age);
  hipLaunchKernelGGL(k6_apply, dim3(1), dim3(256), 0, stream,
                     qn, mem_keys, label, top1, corr, oldest, o_keys, o_vals, o_age);
}
